// SelfAttention_19292993094139
// MI455X (gfx1250) — matrix operand order repaired and checked by reading, not compiled, not run on hardware
//
#include <hip/hip_runtime.h>

// Self-attention block for MI455X (gfx1250, wave32, WMMA 16x16x32 bf16).
// b=4, c=256, h=w=64 -> n=4096, groups=32.
//
// Data layout strategy: every tensor consumed as a WMMA B-matrix is stored
// N-major / K-contiguous (xnT, qT, kT, oT = [b, n, c]) so B fragments are
// 2x ds_load_b128 and all staging moves 16-byte chunks. A-side tensors
// (w_qkv, w_out, v) are already K-contiguous row-major.

#define CH     256
#define NT     4096
#define GROUPS 32
#define SCALE  0.0625f   // 1/sqrt(256)

typedef __attribute__((ext_vector_type(16))) __bf16         v16bf;
typedef __attribute__((ext_vector_type(16))) unsigned short v16u;
typedef __attribute__((ext_vector_type(8)))  unsigned short u16x8;
typedef __attribute__((ext_vector_type(4)))  unsigned short u16x4;
typedef __attribute__((ext_vector_type(8)))  float          v8f;

__device__ __forceinline__ unsigned short f2bf(float f) {
  unsigned int u = __float_as_uint(f);
  u += 0x7fffu + ((u >> 16) & 1u);          // round-to-nearest-even
  return (unsigned short)(u >> 16);
}

__device__ __forceinline__ v8f wmma_bf16(v16u a, v16u b, v8f c) {
  return __builtin_amdgcn_wmma_f32_16x16x32_bf16(
      false, __builtin_bit_cast(v16bf, a),
      false, __builtin_bit_cast(v16bf, b),
      (short)0, c, false, false);
}

// A fragment: 16(M) x 32(K), row-major (K-contiguous) tile, leading dim ld.
// Lane L holds row M=L%16; elem e -> K=(L/16)*16+e.
__device__ __forceinline__ v16u ld_frag_a(const unsigned short* p, int ld, int lane) {
  const int half = (lane >> 4) & 1;
  const unsigned short* r = p + (size_t)(lane & 15) * ld;
  v16u f;
#pragma unroll
  for (int e = 0; e < 8; ++e) f[e] = r[half * 16 + e];
#pragma unroll
  for (int e = 0; e < 8; ++e) f[8 + e] = r[half * 16 + 8 + e];
  return f;
}

// B fragment: 32(K) x 16(N) taken from an N-major tile (rows = N, K contiguous).
// Lane L holds col N=L%16; elem e -> K = (L/16)*16 + e  ==> 16 contiguous u16.
__device__ __forceinline__ v16u ld_frag_bt(const unsigned short* p, int ld, int lane) {
  const int half = (lane >> 4) & 1;
  const unsigned short* r = p + (size_t)(lane & 15) * ld + half * 16;
  v16u f;
#pragma unroll
  for (int e = 0; e < 16; ++e) f[e] = r[e];
  return f;
}

// --------------------------------------------------------------- GroupNorm
// x[b,c,n] (f32) -> xnT[b,n,c] (bf16), transposed via an LDS tile so global
// reads stay coalesced and writes are 16B chunks.
__global__ __launch_bounds__(256) void gn_kernel(const float* __restrict__ x,
                                                 const float* __restrict__ gamma,
                                                 const float* __restrict__ beta,
                                                 unsigned short* __restrict__ xnT) {
  const int b = blockIdx.x >> 5, g = blockIdx.x & 31;
  const size_t base = ((size_t)b * CH + (size_t)g * 8) * NT;
  const int M = 8 * NT;                      // 32768 elements per group
  float s = 0.f, s2 = 0.f;
  for (int i = threadIdx.x; i < M; i += 256) {
    float t = x[base + i]; s += t; s2 += t * t;
  }
  __shared__ float rs[256], rq[256];
  rs[threadIdx.x] = s; rq[threadIdx.x] = s2;
  __syncthreads();
  for (int off = 128; off > 0; off >>= 1) {
    if (threadIdx.x < off) { rs[threadIdx.x] += rs[threadIdx.x + off];
                             rq[threadIdx.x] += rq[threadIdx.x + off]; }
    __syncthreads();
  }
  const float mean = rs[0] / (float)M;
  const float var  = rq[0] / (float)M - mean * mean;
  const float rstd = rsqrtf(var + 1e-5f);
  __shared__ float sco[8], ssh[8];
  if (threadIdx.x < 8) {
    const int c = g * 8 + threadIdx.x;
    const float sc = rstd * gamma[c];
    sco[threadIdx.x] = sc;
    ssh[threadIdx.x] = beta[c] - mean * sc;
  }
  __shared__ float tile[8 * 256];
  for (int nt = 0; nt < NT / 256; ++nt) {
    const int n0 = nt * 256;
    __syncthreads();
    for (int ii = threadIdx.x; ii < 8 * 256; ii += 256) {
      const int cl = ii >> 8, n = ii & 255;
      tile[cl * 256 + n] = x[base + (size_t)cl * NT + n0 + n];
    }
    __syncthreads();
    const int n = threadIdx.x;
    u16x8 pk;
#pragma unroll
    for (int cl = 0; cl < 8; ++cl)
      pk[cl] = f2bf(tile[cl * 256 + n] * sco[cl] + ssh[cl]);
    *(u16x8*)&xnT[((size_t)b * NT + n0 + n) * CH + g * 8] = pk;
  }
}

// -------------------------------------------------------------- QKV GEMM
// qkv[o,n] = sum_c wqkv[o,c]*xnT[n,c] + bqkv[o]; scatter to qT / kT / v.
__global__ __launch_bounds__(256) void qkv_kernel(const unsigned short* __restrict__ xnT,
                                                  const float* __restrict__ wqkv,
                                                  const float* __restrict__ bqkv,
                                                  unsigned short* __restrict__ qT,
                                                  unsigned short* __restrict__ kT,
                                                  unsigned short* __restrict__ v) {
  const int n0 = blockIdx.x * 64, m0 = blockIdx.y * 64, b = blockIdx.z;
  __shared__ __attribute__((aligned(16))) unsigned short sA[64 * 40];  // [m][k]
  __shared__ __attribute__((aligned(16))) unsigned short sB[64 * 40];  // [n][k]
  const int tid = threadIdx.x, w = tid >> 5, lane = tid & 31;
  const int ln = lane & 15, half = lane >> 4;
  const int jsub = w & 3, mbase = (w >> 2) * 32;
  v8f acc[2];
#pragma unroll
  for (int t = 0; t < 2; ++t)
#pragma unroll
    for (int r = 0; r < 8; ++r) acc[t][r] = 0.f;

  for (int ks = 0; ks < 8; ++ks) {
    const int k0 = ks * 32;
    // A: 64x32 f32 -> bf16, 16B-wide reads, 8B-wide LDS stores
    for (int idx = tid; idx < 64 * 8; idx += 256) {
      const int r = idx >> 3, cc = (idx & 7) * 4;
      const float4 f = *(const float4*)&wqkv[(size_t)(m0 + r) * CH + k0 + cc];
      u16x4 pk = { f2bf(f.x), f2bf(f.y), f2bf(f.z), f2bf(f.w) };
      *(u16x4*)&sA[r * 40 + cc] = pk;
    }
    // B: 64 rows (n) x 32 (k), one 16B chunk x 4 per row
    for (int idx = tid; idx < 64 * 4; idx += 256) {
      const int n = idx >> 2, kk = (idx & 3) * 8;
      *(u16x8*)&sB[n * 40 + kk] =
          *(const u16x8*)&xnT[((size_t)b * NT + n0 + n) * CH + k0 + kk];
    }
    __syncthreads();
    v16u bf = ld_frag_bt(&sB[(size_t)(jsub * 16) * 40], 40, lane);
    v16u a0 = ld_frag_a(&sA[(size_t)mbase * 40], 40, lane);
    v16u a1 = ld_frag_a(&sA[(size_t)(mbase + 16) * 40], 40, lane);
    acc[0] = wmma_bf16(a0, bf, acc[0]);
    acc[1] = wmma_bf16(a1, bf, acc[1]);
    __syncthreads();
  }
  const int col = n0 + jsub * 16 + ln;
  if (m0 < 2 * CH) {  // q or k region: packed 16B stores into [n][c] layout
    unsigned short* dst = (m0 < CH) ? &qT[((size_t)b * NT + col) * CH]
                                    : &kT[((size_t)b * NT + col) * CH - CH];
#pragma unroll
    for (int t = 0; t < 2; ++t) {
      const int row0 = m0 + mbase + t * 16 + 8 * half;
      u16x8 pk;
#pragma unroll
      for (int r = 0; r < 8; ++r) pk[r] = f2bf(acc[t][r] + bqkv[row0 + r]);
      *(u16x8*)&dst[row0] = pk;
    }
  } else {            // v region: [c][n] layout (consumed as A-matrix)
#pragma unroll
    for (int t = 0; t < 2; ++t)
#pragma unroll
      for (int r = 0; r < 8; ++r) {
        const int row = m0 + mbase + t * 16 + r + 8 * half;
        v[((size_t)b * CH + (row - 2 * CH)) * NT + col] = f2bf(acc[t][r] + bqkv[row]);
      }
  }
}

// -------------------------------------------------------- Flash attention
// S[i,j] = sum_c kT[i,c] qT[j,c] * scale ; softmax over i ; O[c,j] = sum_i v[c,i] P[i,j]
#define JB   64          // query (j) block
#define IT   32          // key (i) tile
#define SSLD 36          // sS [j][i] leading dim (f32)
#define PTLD 40          // sP [j][i] leading dim (u16)
#define KLD  (CH + 8)    // 264  kT tile [i][c]
#define VLD  (IT + 8)    // 40   v tile  [c][i]
#define QLD  (CH + 8)    // 264  qT tile [j][c]

__global__ __launch_bounds__(256) void attn_kernel(const unsigned short* __restrict__ qT,
                                                   const unsigned short* __restrict__ kT,
                                                   const unsigned short* __restrict__ v,
                                                   unsigned short* __restrict__ oT) {
  const int j0 = blockIdx.x * JB, b = blockIdx.y;
  const int tid = threadIdx.x, w = tid >> 5, lane = tid & 31;
  const int ln = lane & 15, half = lane >> 4;
  const int jsub = w & 3;              // 0..3 : 16-column strip
  const int isub = w >> 2;             // 0..1 : key-row half (S phase)
  const int c0   = (w >> 2) * 128;     // channel half (O phase)

  __shared__ __attribute__((aligned(16))) float          sS[JB * SSLD];  // scores [j][i]
  __shared__ __attribute__((aligned(16))) unsigned short sP[JB * PTLD]; // probs  [j][i]
  __shared__ float sM[JB], sL[JB], sAl[JB];
  __shared__ __attribute__((aligned(16))) unsigned short sT[IT * KLD + CH * VLD];
  unsigned short* sK = sT;                              // [IT][KLD]
  unsigned short* sV = sT + IT * KLD;                   // [CH][VLD]
  unsigned short* sQ = sT;                              // init overlay [JB][QLD]

  // ---- load Q block once (16B chunks), hoist B-fragments into registers
  for (int idx = tid; idx < JB * (CH / 8); idx += 256) {
    const int j = idx >> 5, cc = (idx & 31) * 8;
    *(u16x8*)&sQ[(size_t)j * QLD + cc] =
        *(const u16x8*)&qT[((size_t)b * NT + j0 + j) * CH + cc];
  }
  if (tid < JB) { sM[tid] = -1e30f; sL[tid] = 0.f; }
  __syncthreads();
  v16u qf[8];
#pragma unroll
  for (int kk = 0; kk < 8; ++kk)
    qf[kk] = ld_frag_bt(&sQ[(size_t)(jsub * 16) * QLD + kk * 32], QLD, lane);
  __syncthreads();   // done with sQ overlay

  v8f accO[8];
#pragma unroll
  for (int t = 0; t < 8; ++t)
#pragma unroll
    for (int r = 0; r < 8; ++r) accO[t][r] = 0.f;

  for (int it = 0; it < NT / IT; ++it) {
    const int i0 = it * IT;
    // ---- stage K^T and V tiles (16B chunks)
    for (int idx = tid; idx < IT * (CH / 8); idx += 256) {
      const int i = idx >> 5, cc = (idx & 31) * 8;
      *(u16x8*)&sK[(size_t)i * KLD + cc] =
          *(const u16x8*)&kT[((size_t)b * NT + i0 + i) * CH + cc];
    }
    for (int idx = tid; idx < CH * (IT / 8); idx += 256) {
      const int c = idx >> 2, ii = (idx & 3) * 8;
      *(u16x8*)&sV[(size_t)c * VLD + ii] =
          *(const u16x8*)&v[((size_t)b * CH + c) * NT + i0 + ii];
    }
    __syncthreads();

    // ---- S = K^T Q  (each wave: one 16x16 tile, K=256 over 8 WMMAs)
    v8f accS;
#pragma unroll
    for (int r = 0; r < 8; ++r) accS[r] = 0.f;
#pragma unroll
    for (int kk = 0; kk < 8; ++kk) {
      v16u ka = ld_frag_a(&sK[(size_t)(isub * 16) * KLD + kk * 32], KLD, lane);
      accS = wmma_bf16(ka, qf[kk], accS);
    }
    // store transposed: [j][i], 8 contiguous f32 per lane
    {
      float* srow = &sS[(size_t)(jsub * 16 + ln) * SSLD + isub * 16 + 8 * half];
#pragma unroll
      for (int r = 0; r < 8; ++r) srow[r] = accS[r];
    }
    __syncthreads();

    // ---- online softmax per query column (rows of sS are contiguous)
    if (tid < JB) {
      const int j = tid;
      const float* srow = &sS[j * SSLD];
      float mold = sM[j], mnew = mold;
#pragma unroll 8
      for (int i = 0; i < IT; ++i) mnew = fmaxf(mnew, srow[i] * SCALE);
      const float alpha = __expf(mold - mnew);
      float lsum = 0.f;
      unsigned short* prow = &sP[j * PTLD];
#pragma unroll 8
      for (int i = 0; i < IT; ++i) {
        const float p = __expf(srow[i] * SCALE - mnew);
        prow[i] = f2bf(p);
        lsum += p;
      }
      sL[j] = sL[j] * alpha + lsum;
      sM[j] = mnew;
      sAl[j] = alpha;
    }
    __syncthreads();

    // ---- O = O*alpha + V P  (each wave: 128 channels x 16 cols, K=32)
    const float alpha = sAl[jsub * 16 + ln];
#pragma unroll
    for (int t = 0; t < 8; ++t)
#pragma unroll
      for (int r = 0; r < 8; ++r) accO[t][r] *= alpha;
    v16u pb = ld_frag_bt(&sP[(size_t)(jsub * 16) * PTLD], PTLD, lane);
#pragma unroll
    for (int t = 0; t < 8; ++t) {
      v16u va = ld_frag_a(&sV[(size_t)(c0 + t * 16) * VLD], VLD, lane);
      accO[t] = wmma_bf16(va, pb, accO[t]);
    }
    __syncthreads();
  }

  // ---- normalize and store to oT[b,n,c] as packed 16B chunks
  const float linv = 1.0f / sL[jsub * 16 + ln];
  const int col = j0 + jsub * 16 + ln;
  unsigned short* dst = &oT[((size_t)b * NT + col) * CH];
#pragma unroll
  for (int t = 0; t < 8; ++t) {
    u16x8 pk;
#pragma unroll
    for (int r = 0; r < 8; ++r) pk[r] = f2bf(accO[t][r] * linv);
    *(u16x8*)&dst[c0 + t * 16 + 8 * half] = pk;
  }
}

// ------------------------------------------------------ Output projection
__global__ __launch_bounds__(256) void outproj_kernel(const unsigned short* __restrict__ oT,
                                                      const float* __restrict__ wout,
                                                      const float* __restrict__ bout,
                                                      float* __restrict__ out) {
  const int n0 = blockIdx.x * 64, m0 = blockIdx.y * 64, b = blockIdx.z;
  __shared__ __attribute__((aligned(16))) unsigned short sA[64 * 40];
  __shared__ __attribute__((aligned(16))) unsigned short sB[64 * 40];
  const int tid = threadIdx.x, w = tid >> 5, lane = tid & 31;
  const int ln = lane & 15, half = lane >> 4;
  const int jsub = w & 3, mbase = (w >> 2) * 32;
  v8f acc[2];
#pragma unroll
  for (int t = 0; t < 2; ++t)
#pragma unroll
    for (int r = 0; r < 8; ++r) acc[t][r] = 0.f;

  for (int ks = 0; ks < 8; ++ks) {
    const int k0 = ks * 32;
    for (int idx = tid; idx < 64 * 8; idx += 256) {
      const int r = idx >> 3, cc = (idx & 7) * 4;
      const float4 f = *(const float4*)&wout[(size_t)(m0 + r) * CH + k0 + cc];
      u16x4 pk = { f2bf(f.x), f2bf(f.y), f2bf(f.z), f2bf(f.w) };
      *(u16x4*)&sA[r * 40 + cc] = pk;
    }
    for (int idx = tid; idx < 64 * 4; idx += 256) {
      const int n = idx >> 2, kk = (idx & 3) * 8;
      *(u16x8*)&sB[n * 40 + kk] =
          *(const u16x8*)&oT[((size_t)b * NT + n0 + n) * CH + k0 + kk];
    }
    __syncthreads();
    v16u bf = ld_frag_bt(&sB[(size_t)(jsub * 16) * 40], 40, lane);
    v16u a0 = ld_frag_a(&sA[(size_t)mbase * 40], 40, lane);
    v16u a1 = ld_frag_a(&sA[(size_t)(mbase + 16) * 40], 40, lane);
    acc[0] = wmma_bf16(a0, bf, acc[0]);
    acc[1] = wmma_bf16(a1, bf, acc[1]);
    __syncthreads();
  }
  const int col = n0 + jsub * 16 + ln;
#pragma unroll
  for (int t = 0; t < 2; ++t)
#pragma unroll
    for (int r = 0; r < 8; ++r) {
      const int row = m0 + mbase + t * 16 + r + 8 * half;
      out[((size_t)b * CH + row) * NT + col] = acc[t][r] + bout[row];
    }
}

// ----------------------------------------------------------------- launch
extern "C" void kernel_launch(void* const* d_in, const int* in_sizes, int n_in,
                              void* d_out, int out_size, void* d_ws, size_t ws_size,
                              hipStream_t stream) {
  (void)in_sizes; (void)n_in; (void)out_size; (void)ws_size;
  const float* x     = (const float*)d_in[0];
  const float* gamma = (const float*)d_in[1];
  const float* beta  = (const float*)d_in[2];
  const float* wqkv  = (const float*)d_in[3];
  const float* bqkv  = (const float*)d_in[4];
  const float* wout  = (const float*)d_in[5];
  const float* bout  = (const float*)d_in[6];
  float* out = (float*)d_out;

  // Workspace (bf16 as u16): 5 buffers of 4*256*4096 elements (~40 MB total)
  const size_t SZ = (size_t)4 * CH * NT;
  unsigned short* ws  = (unsigned short*)d_ws;
  unsigned short* xnT = ws;             // [b, n, c]
  unsigned short* qT  = ws + SZ;        // [b, n, c]
  unsigned short* kT  = ws + 2 * SZ;    // [b, n, c]
  unsigned short* v   = ws + 3 * SZ;    // [b, c, n]
  unsigned short* oT  = ws + 4 * SZ;    // [b, n, c]

  gn_kernel<<<dim3(4 * GROUPS), 256, 0, stream>>>(x, gamma, beta, xnT);
  qkv_kernel<<<dim3(NT / 64, (3 * CH) / 64, 4), 256, 0, stream>>>(xnT, wqkv, bqkv, qT, kT, v);
  attn_kernel<<<dim3(NT / JB, 4), 256, 0, stream>>>(qT, kT, v, oT);
  outproj_kernel<<<dim3(NT / 64, CH / 64, 4), 256, 0, stream>>>(oT, wout, bout, out);
}
